// STFT_44607530336505
// MI455X (gfx1250) — matrix operand order repaired and checked by reading, not compiled, not run on hardware
//
#include <hip/hip_runtime.h>
#include <math.h>

// ---------------- problem constants ----------------
#define FL      1024
#define HOP     256
#define CUT     513                 // FILTER_LENGTH/2 + 1
#define BATCH   16
#define NS      1000000
#define FRAMES  3907                // (NS / HOP) + 1 with reflect pad 512
#define PADLEN  (NS + FL)           // 1001024

// ---------------- tiling ----------------
#define MT          33              // ceil(513 / 16) frequency tiles
#define KC          32              // 1024 / 32 k-chunks
#define NFR         256             // frames per workgroup (8 waves x 32)
#define NT          16              // ceil(3907 / 256)
#define WIN_H       ((NFR - 1) * HOP + FL)  // 66304 halves of signal window
#define WIN_B       (WIN_H * 2)             // 132608 bytes LDS
#define AFRAG_H     (32 * 16)               // halves per (kc) A-fragment (all lanes)
#define PACKA_H     (2 * MT * KC * 32 * 16) // 1,081,344 halves = 2,162,688 B
#define SIG_OFF_B   ((size_t)PACKA_H * 2)   // 256B aligned
#define WS_NEEDED   (SIG_OFF_B + (size_t)BATCH * PADLEN * 2)

typedef __attribute__((ext_vector_type(16))) _Float16 v16h;
typedef __attribute__((ext_vector_type(4)))  _Float16 v4h;
typedef __attribute__((ext_vector_type(8)))  float    v8f;

// ---------------------------------------------------------------------------
// Pass 1: pack basis (f32, rows 0..512 = real, 513..1025 = imag) into f16
// A-fragments in contiguous-k per-lane order (B operand uses the same order):
//   lane<16 : j=0..15 -> K=j     (rows M = lane)
//   lane>=16: j=0..15 -> K=j+16  (rows M = lane-16)
// Layout: [mat(2)][mtile(33)][kc(32)][lane(32)][j(16)] halves.
// ---------------------------------------------------------------------------
__global__ void stft_pack_basis(const float* __restrict__ basis,
                                _Float16* __restrict__ packA) {
    int g = blockIdx.x * blockDim.x + threadIdx.x;
    const int total = 2 * MT * KC * 32;
    if (g >= total) return;
    int lane = g & 31;
    int t    = g >> 5;
    int kc   = t % KC;  t /= KC;
    int mtl  = t % MT;
    int mat  = t / MT;
    int freq = mtl * 16 + (lane & 15);
    bool valid = freq < CUT;
    int row = (mat == 0) ? freq : (CUT + freq);
    v16h h;
#pragma unroll
    for (int j = 0; j < 16; ++j) {
        int kk = (lane < 16) ? j : (j + 16);
        kk += kc * 32;
        float v = valid ? basis[row * FL + kk] : 0.0f;
        h[j] = (_Float16)v;
    }
    *(v16h*)(packA + (size_t)g * 16) = h;
}

// ---------------------------------------------------------------------------
// Pass 2: materialize reflect-padded signal as f16 (one copy, L2-resident).
// ---------------------------------------------------------------------------
__global__ void stft_pack_signal(const float* __restrict__ x,
                                 _Float16* __restrict__ sig) {
    long long g = (long long)blockIdx.x * blockDim.x + threadIdx.x;
    const long long total = (long long)BATCH * (PADLEN / 4);
    if (g >= total) return;
    int b  = (int)(g / (PADLEN / 4));
    int p4 = (int)(g % (PADLEN / 4)) * 4;
    const float* xb = x + (long long)b * NS;
    v4h h;
#pragma unroll
    for (int k = 0; k < 4; ++k) {
        int idx = p4 + k - (FL / 2);
        if (idx < 0)   idx = -idx;                 // left reflect
        if (idx >= NS) idx = 2 * (NS - 1) - idx;   // right reflect
        h[k] = (_Float16)xb[idx];
    }
    *(v4h*)(sig + (long long)b * PADLEN + p4) = h;
}

// ---------------------------------------------------------------------------
// Main: per WG = 16 freqs x 256 frames x 1 batch; 8 waves, each 2 N-subtiles.
// K loop: 32 chunks of 32; per chunk 4 x v_wmma_f32_16x16x32_f16
// (real/imag x 2 frame-subtiles, A fragments reused across subtiles).
// Window staged via GLOBAL_LOAD_ASYNC_TO_LDS_B128 (ASYNCcnt path).
// ---------------------------------------------------------------------------
__global__ void __launch_bounds__(256)
stft_wmma(const _Float16* __restrict__ packA,
          const _Float16* __restrict__ sig,
          float* __restrict__ out) {
    extern __shared__ __align__(16) _Float16 swin[];   // WIN_H halves

    const int tid   = threadIdx.x;
    const int b     = blockIdx.z;
    const int mtl   = blockIdx.y;
    const int fb    = blockIdx.x * NFR;
    const int wbase = fb * HOP;                        // window start (padded idx)
    const _Float16* sb = sig + (long long)b * PADLEN;

    // ---- stage signal window into LDS via async global->LDS (no VGPR data) ----
    // Window positions >= PADLEN only feed frames >= 3907 (masked in epilogue),
    // so clamp the global address instead of branching: EXEC stays all-ones.
    {
        const int NV = WIN_B / 16;                     // 8288 x 16B granules
        for (int i = tid; i < NV; i += 256) {
            int g = wbase + i * 8;                     // halves
            if (g + 8 > PADLEN) g = PADLEN - 8;        // clamp (masked frames only)
            unsigned lds_off = (unsigned)(unsigned long long)(swin) + (unsigned)(i * 16);
            unsigned goff    = (unsigned)(g * 2);      // byte offset from batch base
            asm volatile("global_load_async_to_lds_b128 %0, %1, %2"
                         :: "v"(lds_off), "v"(goff), "s"(sb)
                         : "memory");
        }
        asm volatile("s_wait_asynccnt 0x0" ::: "memory");
    }
    __syncthreads();

    const int w     = tid >> 5;        // wave id 0..7
    const int lane  = tid & 31;
    const int nloc  = lane & 15;       // N index within 16x16 tile
    const int khalf = (lane < 16) ? 0 : 16;   // dense 16-bit B: K half per lane grp

    v8f accR0 = {}; v8f accI0 = {};
    v8f accR1 = {}; v8f accI1 = {};

    const _Float16* aR  = packA + ((size_t)(0 * MT + mtl) * KC) * AFRAG_H + lane * 16;
    const _Float16* aI  = packA + ((size_t)(1 * MT + mtl) * KC) * AFRAG_H + lane * 16;
    const _Float16* bp0 = swin + (w * 32 + nloc) * HOP + khalf;
    const _Float16* bp1 = bp0 + 16 * HOP;

#pragma unroll 2
    for (int kc = 0; kc < KC; ++kc) {
        v16h afR = *(const v16h*)(aR + (size_t)kc * AFRAG_H);
        v16h afI = *(const v16h*)(aI + (size_t)kc * AFRAG_H);
        v16h bf0 = *(const v16h*)(bp0 + kc * 32);   // 2x ds_load_b128
        v16h bf1 = *(const v16h*)(bp1 + kc * 32);
        accR0 = __builtin_amdgcn_wmma_f32_16x16x32_f16(
                    false, afR, false, bf0, (short)0, accR0, false, false);
        accR1 = __builtin_amdgcn_wmma_f32_16x16x32_f16(
                    false, afR, false, bf1, (short)0, accR1, false, false);
        accI0 = __builtin_amdgcn_wmma_f32_16x16x32_f16(
                    false, afI, false, bf0, (short)0, accI0, false, false);
        accI1 = __builtin_amdgcn_wmma_f32_16x16x32_f16(
                    false, afI, false, bf1, (short)0, accI1, false, false);
    }

    // ---- epilogue: magnitude + phase, non-temporal stores ----
    const int frame0 = fb + w * 32 + nloc;
    const int frame1 = frame0 + 16;
    const int mbase  = (lane < 16) ? 0 : 8;   // C/D layout: VGPR r -> M=r / M=8+r
    const long long PH = (long long)BATCH * CUT * FRAMES;
#pragma unroll
    for (int r = 0; r < 8; ++r) {
        int freq = mtl * 16 + mbase + r;
        if (freq < CUT) {
            long long row = ((long long)b * CUT + freq) * (long long)FRAMES;
            if (frame0 < FRAMES) {
                float re  = accR0[r];
                float im  = accI0[r];
                __builtin_nontemporal_store(sqrtf(re * re + im * im), out + row + frame0);
                __builtin_nontemporal_store(atan2f(im, re),           out + PH + row + frame0);
            }
            if (frame1 < FRAMES) {
                float re  = accR1[r];
                float im  = accI1[r];
                __builtin_nontemporal_store(sqrtf(re * re + im * im), out + row + frame1);
                __builtin_nontemporal_store(atan2f(im, re),           out + PH + row + frame1);
            }
        }
    }
}

extern "C" void kernel_launch(void* const* d_in, const int* in_sizes, int n_in,
                              void* d_out, int out_size, void* d_ws, size_t ws_size,
                              hipStream_t stream) {
    const float* x     = (const float*)d_in[0];   // (16, 1000000) f32
    const float* basis = (const float*)d_in[1];   // (1026, 1, 1024) f32
    float* out = (float*)d_out;                   // mag (16,513,3907) ++ phase

    if (ws_size < WS_NEEDED) return;              // needs ~34.2 MB scratch
    _Float16* packA = (_Float16*)d_ws;
    _Float16* sigp  = (_Float16*)((char*)d_ws + SIG_OFF_B);

    {   // pack basis fragments: 2*33*32*32 lane-groups
        int total  = 2 * MT * KC * 32;
        int blocks = (total + 255) / 256;
        stft_pack_basis<<<blocks, 256, 0, stream>>>(basis, packA);
    }
    {   // pack reflect-padded f16 signal
        long long total = (long long)BATCH * (PADLEN / 4);
        int blocks = (int)((total + 255) / 256);
        stft_pack_signal<<<blocks, 256, 0, stream>>>(x, sigp);
    }
    {   // main WMMA STFT
        dim3 grid(NT, MT, BATCH);                 // 16 x 33 x 16 = 8448 WGs
        stft_wmma<<<grid, 256, (size_t)WIN_B, stream>>>(packA, sigp, out);
    }
}
